// LlamaMLP_85538568667280
// MI455X (gfx1250) — 8-bit matrix operand order repaired and checked by reading, compiled, not run on hardware
//
#include <hip/hip_runtime.h>
#include <cstdint>
#include <cstddef>

#define S_DIM 4096
#define H_DIM 4096
#define I_DIM 14336
#define I2_DIM (2 * I_DIM)

#define KT1 (H_DIM / 64)
#define KT2 (I_DIM / 64)
#define MT  (S_DIM / 16)
#define NT1 (I_DIM / 16)
#define NT2 (H_DIM / 16)

typedef int v8i __attribute__((ext_vector_type(8)));

__device__ __forceinline__ size_t a_pack_off(size_t tile, int M, int kk) {
  int group  = kk >> 4;
  int within = kk & 15;
  int half   = within >> 3;
  int sub    = within & 7;
  int r      = group * 2 + (sub >> 2);
  int b      = sub & 3;
  int lane   = M + 16 * half;
  return tile * 1024 + (size_t)lane * 32 + r * 4 + b;
}

__device__ __forceinline__ size_t b_pack_off(size_t tile, int N, int kk) {
  int hi   = kk >> 4;
  int mid  = (kk >> 3) & 1;
  int low  = kk & 7;
  int r    = hi * 2 + (low >> 2);
  int b    = low & 3;
  int lane = N + 16 * mid;
  return tile * 1024 + (size_t)lane * 32 + r * 4 + b;
}

__global__ __launch_bounds__(256) void pack_qx_kernel(
    const int* __restrict__ qx, unsigned char* __restrict__ Apack,
    float* __restrict__ xsumf) {
  int row = blockIdx.x, tid = threadIdx.x;
  int mt = row >> 4, M = row & 15;
  const int* qrow = qx + (size_t)row * H_DIM;
  int local = 0;
  for (int j = 0; j < H_DIM / 256; ++j) {
    int k = j * 256 + tid;
    int v = qrow[k];
    local += v;
    Apack[a_pack_off((size_t)mt * KT1 + (k >> 6), M, k & 63)] =
        (unsigned char)(signed char)v;
  }
  __shared__ int red[256];
  red[tid] = local;
  __syncthreads();
  for (int s = 128; s > 0; s >>= 1) {
    if (tid < s) red[tid] += red[tid + s];
    __syncthreads();
  }
  if (tid == 0) xsumf[row] = (float)red[0];
}

__global__ __launch_bounds__(256) void pack_w_kernel(
    const float* __restrict__ W, unsigned char* __restrict__ Bpack,
    int Ncols, int K, int kt_count) {
  size_t total = (size_t)Ncols * (size_t)K;
  for (size_t idx = (size_t)blockIdx.x * blockDim.x + threadIdx.x; idx < total;
       idx += (size_t)gridDim.x * blockDim.x) {
    int n = (int)(idx / (size_t)K);
    int k = (int)(idx - (size_t)n * K);
    unsigned char v = (unsigned char)(int)W[idx];
    Bpack[b_pack_off((size_t)(n >> 4) * kt_count + (k >> 6), n & 15, k & 63)] = v;
  }
}

__global__ __launch_bounds__(128) void gemm1_silu_kernel(
    const unsigned char* __restrict__ Apack,
    const unsigned char* __restrict__ Bpack,
    const float* __restrict__ x_scale, const float* __restrict__ xsumf,
    const float* __restrict__ gu_ws, const float* __restrict__ gu_wz,
    _Float16* __restrict__ Y, unsigned int* __restrict__ rowmax) {
  int lane = threadIdx.x & 31;
  int wave = threadIdx.x >> 5;
  int nt  = blockIdx.x * 4 + wave;
  int mt0 = blockIdx.y * 4;

  const size_t laneoff = (size_t)lane * 32;
  const unsigned char* Ap = Apack + (size_t)mt0 * KT1 * 1024 + laneoff;
  const unsigned char* Bg = Bpack + (size_t)nt * KT1 * 1024 + laneoff;
  const unsigned char* Bu = Bpack + (size_t)(nt + NT1) * KT1 * 1024 + laneoff;

  v8i accG[4], accU[4];
#pragma unroll
  for (int m = 0; m < 4; ++m) {
    accG[m] = (v8i){0, 0, 0, 0, 0, 0, 0, 0};
    accU[m] = (v8i){0, 0, 0, 0, 0, 0, 0, 0};
  }

#pragma clang loop unroll(disable)
  for (int kt = 0; kt < KT1; ++kt) {
    v8i bg = *(const v8i*)(Bg + (size_t)kt * 1024);
    v8i bu = *(const v8i*)(Bu + (size_t)kt * 1024);
    __builtin_prefetch(Bg + (size_t)(kt + 1) * 1024, 0, 1);
    __builtin_prefetch(Bu + (size_t)(kt + 1) * 1024, 0, 1);
#pragma unroll
    for (int m = 0; m < 4; ++m) {
      v8i a = *(const v8i*)(Ap + ((size_t)m * KT1 + kt) * 1024);
      accG[m] = __builtin_amdgcn_wmma_i32_16x16x64_iu8(true, a, false, bg,
                                                       accG[m], false, false);
      accU[m] = __builtin_amdgcn_wmma_i32_16x16x64_iu8(true, a, false, bu,
                                                       accU[m], false, false);
    }
  }

  int N = lane & 15;
  int rowHalf = (lane >> 4) * 8;
  int gcol = nt * 16 + N;
  int ucol = gcol + I_DIM;
  float wsg = gu_ws[gcol], wzg = gu_wz[gcol];
  float wsu = gu_ws[ucol], wzu = gu_wz[ucol];
#pragma unroll
  for (int m = 0; m < 4; ++m) {
#pragma unroll
    for (int r = 0; r < 8; ++r) {
      int row = (mt0 + m) * 16 + rowHalf + r;
      float xs = x_scale[row], xv = xsumf[row];
      float g = xs * wsg * ((float)accG[m][r] - xv * wzg);
      float u = xs * wsu * ((float)accU[m][r] - xv * wzu);
      float y = (g / (1.0f + expf(-g))) * u;
      Y[(size_t)row * I_DIM + gcol] = (_Float16)y;

      float am = fabsf(y);
#pragma unroll
      for (int s = 1; s < 16; s <<= 1) am = fmaxf(am, __shfl_xor(am, s, 16));
      if (N == 0)
        atomicMax(&rowmax[row], __float_as_uint(am));
    }
  }
}

__global__ __launch_bounds__(256) void quantize_kernel(
    const _Float16* __restrict__ Y, const unsigned int* __restrict__ rowmax,
    float* __restrict__ s2, int* __restrict__ sum2,
    unsigned char* __restrict__ Apack2) {
  int row = blockIdx.x, tid = threadIdx.x;
  float s = __uint_as_float(rowmax[row]) * (1.0f / 127.0f);
  float inv = (s > 0.0f) ? (1.0f / s) : 0.0f;
  int mt = row >> 4, M = row & 15;
  const _Float16* yrow = Y + (size_t)row * I_DIM;
  int local = 0;
  for (int j = 0; j < I_DIM / 256; ++j) {
    int k = j * 256 + tid;
    float q = rintf((float)yrow[k] * inv);
    q = fminf(127.0f, fmaxf(-127.0f, q));
    int qi = (int)q;
    local += qi;
    Apack2[a_pack_off((size_t)mt * KT2 + (k >> 6), M, k & 63)] =
        (unsigned char)(signed char)qi;
  }
  __shared__ int red[256];
  red[tid] = local;
  __syncthreads();
  for (int st = 128; st > 0; st >>= 1) {
    if (tid < st) red[tid] += red[tid + st];
    __syncthreads();
  }
  if (tid == 0) { s2[row] = s; sum2[row] = red[0]; }
}

__global__ __launch_bounds__(128) void gemm2_kernel(
    const unsigned char* __restrict__ Apack,
    const unsigned char* __restrict__ Bpack,
    const float* __restrict__ s2, const int* __restrict__ sum2,
    const float* __restrict__ dn_ws, const float* __restrict__ dn_wz,
    float* __restrict__ out) {
  int lane = threadIdx.x & 31;
  int wave = threadIdx.x >> 5;
  int nt0 = (blockIdx.x * 4 + wave) * 2;
  int mt0 = blockIdx.y * 4;

  const size_t laneoff = (size_t)lane * 32;
  const unsigned char* Ap = Apack + (size_t)mt0 * KT2 * 1024 + laneoff;
  const unsigned char* B0 = Bpack + (size_t)nt0 * KT2 * 1024 + laneoff;
  const unsigned char* B1 = Bpack + (size_t)(nt0 + 1) * KT2 * 1024 + laneoff;

  v8i acc0[4], acc1[4];
#pragma unroll
  for (int m = 0; m < 4; ++m) {
    acc0[m] = (v8i){0, 0, 0, 0, 0, 0, 0, 0};
    acc1[m] = (v8i){0, 0, 0, 0, 0, 0, 0, 0};
  }

#pragma clang loop unroll(disable)
  for (int kt = 0; kt < KT2; ++kt) {
    v8i b0 = *(const v8i*)(B0 + (size_t)kt * 1024);
    v8i b1 = *(const v8i*)(B1 + (size_t)kt * 1024);
    __builtin_prefetch(B0 + (size_t)(kt + 1) * 1024, 0, 1);
    __builtin_prefetch(B1 + (size_t)(kt + 1) * 1024, 0, 1);
#pragma unroll
    for (int m = 0; m < 4; ++m) {
      v8i a = *(const v8i*)(Ap + ((size_t)m * KT2 + kt) * 1024);
      acc0[m] = __builtin_amdgcn_wmma_i32_16x16x64_iu8(true, a, false, b0,
                                                       acc0[m], false, false);
      acc1[m] = __builtin_amdgcn_wmma_i32_16x16x64_iu8(true, a, false, b1,
                                                       acc1[m], false, false);
    }
  }

  int N = lane & 15;
  int rowHalf = (lane >> 4) * 8;
  int col0 = nt0 * 16 + N;
  int col1 = col0 + 16;
  float ws0 = dn_ws[col0], wz0 = dn_wz[col0];
  float ws1 = dn_ws[col1], wz1 = dn_wz[col1];
#pragma unroll
  for (int m = 0; m < 4; ++m) {
#pragma unroll
    for (int r = 0; r < 8; ++r) {
      int row = (mt0 + m) * 16 + rowHalf + r;
      float sc = s2[row];
      float sm = (float)sum2[row];
      out[(size_t)row * H_DIM + col0] =
          sc * ws0 * ((float)acc0[m][r] - sm * wz0);
      out[(size_t)row * H_DIM + col1] =
          sc * ws1 * ((float)acc1[m][r] - sm * wz1);
    }
  }
}

extern "C" void kernel_launch(void* const* d_in, const int* in_sizes, int n_in,
                              void* d_out, int out_size, void* d_ws,
                              size_t ws_size, hipStream_t stream) {
  (void)in_sizes; (void)n_in; (void)out_size; (void)ws_size;
  const int*   qx      = (const int*)d_in[0];
  const float* x_scale = (const float*)d_in[1];
  const float* gu_wq   = (const float*)d_in[2];
  const float* gu_ws   = (const float*)d_in[3];
  const float* gu_wz   = (const float*)d_in[4];
  const float* dn_wq   = (const float*)d_in[5];
  const float* dn_ws   = (const float*)d_in[6];
  const float* dn_wz   = (const float*)d_in[7];
  float* out = (float*)d_out;

  char* ws = (char*)d_ws;
  unsigned char* Apack1 = (unsigned char*)(ws);
  unsigned char* Bpack1 = (unsigned char*)(ws + 16777216ull);
  unsigned char* Apack2 = (unsigned char*)(ws + 134217728ull);
  unsigned char* Bpack2 = (unsigned char*)(ws + 192937984ull);
  _Float16*      Yh     = (_Float16*)(ws + 251658240ull);
  float*        xsumf   = (float*)(ws + 369098752ull);
  unsigned int* rowmax  = (unsigned int*)(ws + 369115136ull);
  float*        s2      = (float*)(ws + 369131520ull);
  int*          sum2    = (int*)(ws + 369147904ull);

  hipMemsetAsync(rowmax, 0, S_DIM * sizeof(unsigned int), stream);

  pack_qx_kernel<<<S_DIM, 256, 0, stream>>>(qx, Apack1, xsumf);
  pack_w_kernel<<<49152, 256, 0, stream>>>(gu_wq, Bpack1, I2_DIM, H_DIM, KT1);
  pack_w_kernel<<<24576, 256, 0, stream>>>(dn_wq, Bpack2, H_DIM, I_DIM, KT2);

  gemm1_silu_kernel<<<dim3(NT1 / 4, MT / 4), 128, 0, stream>>>(
      Apack1, Bpack1, x_scale, xsumf, gu_ws, gu_wz, Yh, rowmax);

  quantize_kernel<<<S_DIM, 256, 0, stream>>>(Yh, rowmax, s2, sum2, Apack2);

  gemm2_kernel<<<dim3(NT2 / 8, MT / 4), 128, 0, stream>>>(
      Apack2, Bpack2, s2, sum2, dn_ws, dn_wz, out);
}
